// MultiHeadAttention_1606317769145
// MI455X (gfx1250) — matrix operand order repaired and checked by reading, not compiled, not run on hardware
//
#include <hip/hip_runtime.h>

// MHA: B=4, T=2048, D=768, H=12, DH=64, causal softmax, SCALE = 1/8
#define Bz 4
#define Tz 2048
#define Dz 768
#define Hz 12
#define DHz 64
#define SCALEz 0.125f
#define NEGINF (-1e30f)

typedef __attribute__((ext_vector_type(16))) _Float16 v16h;
typedef __attribute__((ext_vector_type(8)))  float    v8f;
typedef __attribute__((ext_vector_type(4)))  int      v4i;
typedef unsigned int u32;

union U16x { v16h v; u32 u[8]; };

#if defined(__has_builtin)
#if __has_builtin(__builtin_amdgcn_global_load_async_to_lds_b128)
#define HAVE_ASYNC 1
#endif
#endif

#define AS1 __attribute__((address_space(1)))
#define AS3 __attribute__((address_space(3)))

// 16-byte global -> LDS copy. Async (ASYNCcnt-tracked, no VGPR round trip) when
// the gfx1250 builtin is available; otherwise a synchronous load+store.
// Builtin signature (from hipcc diagnostic): (int4 AS1*, int4 AS3*, imm, imm).
__device__ inline void async_cp16(const void* g, void* l) {
#ifdef HAVE_ASYNC
  // flat LDS address: low 32 bits are the LDS offset (ISA aperture rule)
  __builtin_amdgcn_global_load_async_to_lds_b128(
      (AS1 v4i*)(size_t)g, (AS3 v4i*)(u32)(size_t)l, 0, 0);
#else
  *(uint4*)l = *(const uint4*)g;
#endif
}

__device__ inline void wait_async() {
#ifdef HAVE_ASYNC
#if __has_builtin(__builtin_amdgcn_s_wait_asynccnt)
  __builtin_amdgcn_s_wait_asynccnt(0);
#else
  asm volatile("s_wait_asynccnt 0x0" ::: "memory");
#endif
#endif
}

__device__ inline v8f zero8() {
  v8f z = {0.f, 0.f, 0.f, 0.f, 0.f, 0.f, 0.f, 0.f};
  return z;
}

__device__ inline v8f wmma16(v16h a, v16h b, v8f c) {
  // D = A(16x32 f16) * B(32x16 f16) + C(16x16 f32)
  return __builtin_amdgcn_wmma_f32_16x16x32_f16(false, a, false, b, (short)0, c, false, false);
}

// A operand (16x32, MxK) gather from row-major LDS tile (row stride ld, ld even).
__device__ inline v16h ldA(const _Float16* p, int ld) {
  const int lane = threadIdx.x & 31;
  const int hf = lane >> 4, m = lane & 15;
  const u32* row = (const u32*)(p + m * ld);
  U16x r;
#pragma unroll
  for (int q = 0; q < 8; ++q) {
    int k0 = ((q >> 2) << 4) + hf * 8 + ((q & 3) << 1);
    r.u[q] = row[k0 >> 1];
  }
  return r.v;
}

// B operand (32x16, KxN) with B[k][n] = S[n][k], gathered from row-major S tile.
__device__ inline v16h ldB(const _Float16* p, int ld) {
  const int lane = threadIdx.x & 31;
  const int hf = lane >> 4, n = lane & 15;
  const u32* row = (const u32*)(p + n * ld);
  U16x r;
#pragma unroll
  for (int q = 0; q < 8; ++q) r.u[q] = row[((q >> 2) << 3) + hf * 4 + (q & 3)];
  return r.v;
}

// ---------------- Kernel 0: f32 -> f16 conversion (hoisted out of GEMMs) ------
__global__ __launch_bounds__(256) void k_cvt(const float* __restrict__ src,
                                             _Float16* __restrict__ dst, int n4) {
  int i = blockIdx.x * 256 + threadIdx.x;
  if (i < n4) {
    float4 v = ((const float4*)src)[i];
    union { _Float16 h[4]; uint2 u; } r;
    r.h[0] = (_Float16)v.x; r.h[1] = (_Float16)v.y;
    r.h[2] = (_Float16)v.z; r.h[3] = (_Float16)v.w;
    ((uint2*)dst)[i] = r.u;
  }
}

// ---------------- Kernel 1: qkv = x @ Wqkv^T (f16 in/out) ----------------
// Grid (64, 36), block 128 (4 waves). Block tile 128x64, wave tile 32x64.
// Double-buffered LDS filled by async b128 copies.
__global__ __launch_bounds__(128) void k_qkv(const _Float16* __restrict__ X,
                                             const _Float16* __restrict__ W,
                                             _Float16* __restrict__ Qb,
                                             _Float16* __restrict__ Kb,
                                             _Float16* __restrict__ Vb) {
  __shared__ __align__(16) _Float16 Xs[2][128][32];
  __shared__ __align__(16) _Float16 Ws[2][64][32];
  const int t = threadIdx.x, wv = t >> 5, lane = t & 31, hf = lane >> 4, col = lane & 15;
  const int m0 = blockIdx.x * 128, n0 = blockIdx.y * 64;
  v8f acc0[4] = {zero8(), zero8(), zero8(), zero8()};
  v8f acc1[4] = {zero8(), zero8(), zero8(), zero8()};

  auto stage = [&](int buf, int kb) {
#pragma unroll
    for (int j = 0; j < 4; ++j) {               // 128x32 A tile: 512 x 16B
      int seg = t + j * 128;
      int r = seg >> 2, c = (seg & 3) * 8;
      async_cp16(X + (size_t)(m0 + r) * Dz + kb + c, &Xs[buf][r][c]);
    }
#pragma unroll
    for (int j = 0; j < 2; ++j) {               // 64x32 B tile: 256 x 16B
      int seg = t + j * 128;
      int r = seg >> 2, c = (seg & 3) * 8;
      async_cp16(W + (size_t)(n0 + r) * Dz + kb + c, &Ws[buf][r][c]);
    }
  };

  stage(0, 0);
  wait_async();
  __syncthreads();
  int cur = 0;
  for (int kb = 0; kb < Dz; kb += 32) {
    if (kb + 32 < Dz) stage(cur ^ 1, kb + 32);  // prefetch next tile
    v16h a0 = ldA(&Xs[cur][wv * 32][0], 32);
    v16h a1 = ldA(&Xs[cur][wv * 32 + 16][0], 32);
#pragma unroll
    for (int nt = 0; nt < 4; ++nt) {
      v16h b = ldB(&Ws[cur][nt * 16][0], 32);
      acc0[nt] = wmma16(a0, b, acc0[nt]);
      acc1[nt] = wmma16(a1, b, acc1[nt]);
    }
    wait_async();
    __syncthreads();
    cur ^= 1;
  }

#pragma unroll
  for (int nt = 0; nt < 4; ++nt) {
    int n = n0 + nt * 16 + col;
    _Float16* dst; int nn;
    if (n < Dz)          { dst = Qb; nn = n; }
    else if (n < 2 * Dz) { dst = Kb; nn = n - Dz; }
    else                 { dst = Vb; nn = n - 2 * Dz; }
#pragma unroll
    for (int i = 0; i < 8; ++i) {
      int m = m0 + wv * 32 + hf * 8 + i;
      dst[(size_t)m * Dz + nn] = (_Float16)acc0[nt][i];
      dst[(size_t)(m + 16) * Dz + nn] = (_Float16)acc1[nt][i];
    }
  }
}

// ---------------- Kernel 2: causal flash attention ----------------
// Grid (T/64, B*H), block 128 (4 waves); wave owns 16 queries.
// S^T = K*Q^T so softmax over keys reduces in-lane + one shfl_xor(16).
__global__ __launch_bounds__(128) void k_attn(const _Float16* __restrict__ Qb,
                                              const _Float16* __restrict__ Kb,
                                              const _Float16* __restrict__ Vb,
                                              _Float16* __restrict__ Ob) {
  __shared__ __align__(16) _Float16 Qs[4][16][64];
  __shared__ __align__(16) _Float16 Ks[32][64];
  __shared__ __align__(16) _Float16 Vt[64][32];
  __shared__ __align__(16) _Float16 Ps[4][16][32];
  const int t = threadIdx.x, wv = t >> 5, lane = t & 31, hf = lane >> 4, col = lane & 15;
  const int b = blockIdx.y / Hz, h = blockIdx.y % Hz;
  const int q0 = blockIdx.x * 64;
  const int qw = q0 + wv * 16;
  const size_t hb = (size_t)b * Tz * Dz + (size_t)h * DHz;

  { // per-wave Q tile 16x64 (row-major -> feeds ldB directly), async copy
    const _Float16* qsrc = Qb + hb + (size_t)(qw + col) * Dz + hf * 32;
#pragma unroll
    for (int j = 0; j < 4; ++j)
      async_cp16(qsrc + j * 8, &Qs[wv][col][hf * 32 + j * 8]);
  }

  v8f acc[4] = {zero8(), zero8(), zero8(), zero8()};
  float mrow = NEGINF, lrow = 0.f;
  const int kend = q0 + 64;

  for (int kb = 0; kb < kend; kb += 32) {
    __syncthreads();  // all waves done reading Ks/Vt from previous iteration
    // K tile 32x64: async b128 copies
#pragma unroll
    for (int j = 0; j < 2; ++j) {
      int seg = t + j * 128;
      int r = seg >> 3, c = (seg & 7) * 8;
      async_cp16(Kb + hb + (size_t)(kb + r) * Dz + c, &Ks[r][c]);
    }
    // V tile staged transposed (keeps AV B-operand gathers b32-mergeable)
#pragma unroll
    for (int j = 0; j < 4; ++j) {
      int e = j * 512 + t * 4;
      int r = e >> 6, c = e & 63;
      uint2 vv2 = *(const uint2*)(Vb + hb + (size_t)(kb + r) * Dz + c);
      __align__(8) _Float16 vh[4];
      *(uint2*)vh = vv2;
      Vt[c + 0][r] = vh[0]; Vt[c + 1][r] = vh[1];
      Vt[c + 2][r] = vh[2]; Vt[c + 3][r] = vh[3];
    }
    wait_async();
    __syncthreads();

    // S^T tiles (M=key, N=query), dh reduced in two 32-chunks
    v8f st[2] = {zero8(), zero8()};
#pragma unroll
    for (int s = 0; s < 2; ++s)
#pragma unroll
      for (int c2 = 0; c2 < 2; ++c2) {
        v16h aK = ldA(&Ks[s * 16][c2 * 32], 64);
        v16h bQ = ldB(&Qs[wv][0][c2 * 32], 64);
        st[s] = wmma16(aK, bQ, st[s]);
      }

    // causal mask
    const int qg = qw + col;
#pragma unroll
    for (int s = 0; s < 2; ++s)
#pragma unroll
      for (int i = 0; i < 8; ++i)
        if (kb + s * 16 + hf * 8 + i > qg) st[s][i] = NEGINF;

    // online softmax (per-lane state for query column q = lane&15)
    float tm = NEGINF;
#pragma unroll
    for (int s = 0; s < 2; ++s)
#pragma unroll
      for (int i = 0; i < 8; ++i) tm = fmaxf(tm, st[s][i]);
    tm = fmaxf(tm, __shfl_xor(tm, 16, 32));
    float mnew = fmaxf(mrow, tm);
    float alpha = __expf((mrow - mnew) * SCALEz);

    float pr[2][8]; float rs = 0.f;
#pragma unroll
    for (int s = 0; s < 2; ++s)
#pragma unroll
      for (int i = 0; i < 8; ++i) {
        pr[s][i] = __expf((st[s][i] - mnew) * SCALEz);
        rs += pr[s][i];
      }
    rs += __shfl_xor(rs, 16, 32);
    lrow = lrow * alpha + rs;
    mrow = mnew;

    // rescale O accumulators (rows q = 8*hf+i in O layout)
#pragma unroll
    for (int i = 0; i < 8; ++i) {
      float aq = __shfl(alpha, hf * 8 + i, 32);
#pragma unroll
      for (int nt = 0; nt < 4; ++nt) acc[nt][i] *= aq;
    }

    // P -> per-wave LDS, row-major [q][key0..31]
#pragma unroll
    for (int s = 0; s < 2; ++s)
#pragma unroll
      for (int j = 0; j < 4; ++j) {
        union { _Float16 hh[2]; u32 uu; } pk;
        pk.hh[0] = (_Float16)pr[s][2 * j];
        pk.hh[1] = (_Float16)pr[s][2 * j + 1];
        *(u32*)&Ps[wv][col][s * 16 + hf * 8 + 2 * j] = pk.uu;
      }

    // O += P(16x32) * V(32x64)
    v16h aP = ldA(&Ps[wv][0][0], 32);
#pragma unroll
    for (int nt = 0; nt < 4; ++nt) {
      v16h bV = ldB(&Vt[nt * 16][0], 32);
      acc[nt] = wmma16(aP, bV, acc[nt]);
    }
  }

  // epilogue: divide by row sum, store f16
#pragma unroll
  for (int i = 0; i < 8; ++i) {
    float li = __shfl(lrow, hf * 8 + i, 32);
    float inv = 1.0f / li;
    int tq = qw + hf * 8 + i;
#pragma unroll
    for (int nt = 0; nt < 4; ++nt)
      Ob[hb + (size_t)tq * Dz + nt * 16 + col] = (_Float16)(acc[nt][i] * inv);
  }
}

// ---------------- Kernel 3: out = O @ Wproj^T + bproj (f32 out) ----------------
// Grid (64, 12), block 128. Same double-buffered async structure as k_qkv.
__global__ __launch_bounds__(128) void k_proj(const _Float16* __restrict__ O,
                                              const _Float16* __restrict__ W,
                                              const float* __restrict__ bias,
                                              float* __restrict__ out) {
  __shared__ __align__(16) _Float16 Os[2][128][32];
  __shared__ __align__(16) _Float16 Ws[2][64][32];
  const int t = threadIdx.x, wv = t >> 5, lane = t & 31, hf = lane >> 4, col = lane & 15;
  const int m0 = blockIdx.x * 128, n0 = blockIdx.y * 64;
  v8f acc0[4] = {zero8(), zero8(), zero8(), zero8()};
  v8f acc1[4] = {zero8(), zero8(), zero8(), zero8()};

  auto stage = [&](int buf, int kb) {
#pragma unroll
    for (int j = 0; j < 4; ++j) {
      int seg = t + j * 128;
      int r = seg >> 2, c = (seg & 3) * 8;
      async_cp16(O + (size_t)(m0 + r) * Dz + kb + c, &Os[buf][r][c]);
    }
#pragma unroll
    for (int j = 0; j < 2; ++j) {
      int seg = t + j * 128;
      int r = seg >> 2, c = (seg & 3) * 8;
      async_cp16(W + (size_t)(n0 + r) * Dz + kb + c, &Ws[buf][r][c]);
    }
  };

  stage(0, 0);
  wait_async();
  __syncthreads();
  int cur = 0;
  for (int kb = 0; kb < Dz; kb += 32) {
    if (kb + 32 < Dz) stage(cur ^ 1, kb + 32);
    v16h a0 = ldA(&Os[cur][wv * 32][0], 32);
    v16h a1 = ldA(&Os[cur][wv * 32 + 16][0], 32);
#pragma unroll
    for (int nt = 0; nt < 4; ++nt) {
      v16h b = ldB(&Ws[cur][nt * 16][0], 32);
      acc0[nt] = wmma16(a0, b, acc0[nt]);
      acc1[nt] = wmma16(a1, b, acc1[nt]);
    }
    wait_async();
    __syncthreads();
    cur ^= 1;
  }

#pragma unroll
  for (int nt = 0; nt < 4; ++nt) {
    int n = n0 + nt * 16 + col;
    float bv = bias[n];
#pragma unroll
    for (int i = 0; i < 8; ++i) {
      int m = m0 + wv * 32 + hf * 8 + i;
      out[(size_t)m * Dz + n] = acc0[nt][i] + bv;
      out[(size_t)(m + 16) * Dz + n] = acc1[nt][i] + bv;
    }
  }
}

extern "C" void kernel_launch(void* const* d_in, const int* in_sizes, int n_in,
                              void* d_out, int out_size, void* d_ws, size_t ws_size,
                              hipStream_t stream) {
  const float* x     = (const float*)d_in[0]; // [4,2048,768]
  const float* Wqkv  = (const float*)d_in[1]; // [2304,768]
  const float* Wproj = (const float*)d_in[2]; // [768,768]
  const float* bproj = (const float*)d_in[3]; // [768]
  float* out = (float*)d_out;                 // [4,2048,768] f32

  const size_t NTD  = (size_t)Bz * Tz * Dz;   // 6,291,456
  const size_t NQKV = (size_t)3 * Dz * Dz;    // 1,769,472
  const size_t NPRJ = (size_t)Dz * Dz;        //   589,824

  _Float16* X16 = (_Float16*)d_ws;
  _Float16* Wq16 = X16 + NTD;
  _Float16* Wp16 = Wq16 + NQKV;
  _Float16* Qb = Wp16 + NPRJ;
  _Float16* Kb = Qb + NTD;
  _Float16* Vb = Kb + NTD;
  _Float16* Ob = Vb + NTD;   // total ~68 MB of f16 workspace

  (void)in_sizes; (void)n_in; (void)out_size; (void)ws_size;

  const int n4x = (int)(NTD / 4), n4q = (int)(NQKV / 4), n4p = (int)(NPRJ / 4);
  k_cvt<<<(n4x + 255) / 256, 256, 0, stream>>>(x, X16, n4x);
  k_cvt<<<(n4q + 255) / 256, 256, 0, stream>>>(Wqkv, Wq16, n4q);
  k_cvt<<<(n4p + 255) / 256, 256, 0, stream>>>(Wproj, Wp16, n4p);

  k_qkv <<<dim3((Bz * Tz) / 128, (3 * Dz) / 64), 128, 0, stream>>>(X16, Wq16, Qb, Kb, Vb);
  k_attn<<<dim3(Tz / 64, Bz * Hz),               128, 0, stream>>>(Qb, Kb, Vb, Ob);
  k_proj<<<dim3((Bz * Tz) / 128, Dz / 64),       128, 0, stream>>>(Ob, Wp16, bproj, out);
}
